// PaperInformedMambaLMHeadModel_6562710028721
// MI455X (gfx1250) — compile-verified, matrix operand order checked by reading, not run on hardware
//
#include <hip/hip_runtime.h>
#include <hip/hip_bf16.h>
#include <math.h>

// ---------------------------------------------------------------------------
// Model dims (compile-time constants from the reference)
// ---------------------------------------------------------------------------
#define DMODEL 1024
#define NSTATE 32
#define NLAYER 2
#define VOCAB  32000
#define BSZ    2
#define SEQ    2048
#define BL     (BSZ * SEQ)        // 4096 token rows

// ---------------------------------------------------------------------------
// WMMA types (CDNA5 / gfx1250, wave32)
// ---------------------------------------------------------------------------
typedef __bf16 v16bf __attribute__((ext_vector_type(16)));
typedef __bf16 v2bf  __attribute__((ext_vector_type(2)));
typedef float  v8f   __attribute__((ext_vector_type(8)));

union FragBF16 {
    v16bf v;
    uint4 q[2];   // 2 x 16B  ==  16 bf16
};

// f32 pair -> packed bf16 dword via native v_cvt_pk_bf16_f32
__device__ __forceinline__ unsigned int pack_bf16(float a, float b) {
    v2bf t;
    t[0] = (__bf16)a;
    t[1] = (__bf16)b;
    return __builtin_bit_cast(unsigned int, t);
}

__device__ __forceinline__ unsigned short to_bf16(float a) {
    return __builtin_bit_cast(unsigned short, (__bf16)a);
}

__device__ __forceinline__ float silu(float x) {
    return x / (1.0f + expf(-x));
}

// CDNA5 async global->LDS copy, 16B per lane (ASYNCcnt tracked, no VGPR round
// trip). vdst = per-lane LDS byte address, vaddr = 64-bit global address.
__device__ __forceinline__ void async_copy_b128(void* lds_ptr, const void* gptr) {
    unsigned lds_addr = (unsigned)(size_t)lds_ptr;          // addr[31:0] = LDS offset
    unsigned long long ga = (unsigned long long)(size_t)gptr;
    asm volatile("global_load_async_to_lds_b128 %0, %1, off"
                 :: "v"(lds_addr), "v"(ga) : "memory");
}

__device__ __forceinline__ void wait_async0() {
    asm volatile("s_wait_asynccnt 0x0" ::: "memory");
}
// wait until only the most recent 6 async ops (next tile) are outstanding
__device__ __forceinline__ void wait_async6() {
    asm volatile("s_wait_asynccnt 0x6" ::: "memory");
}

// ---------------------------------------------------------------------------
// Tiled bf16 WMMA GEMM:  C[M,Nc] = A[M,K] @ W[Nc,K]^T (+ bias)
//   A: bf16 row-major [M,K].  W: bf16 row-major [Nc,K].  C: f32, ldc.
// Block: 256 threads = 8 waves (2x4).  Tile: 128M x 256N x 32K.
// Per wave: 64x64 patch = 4x4 accumulators -> 16 v_wmma per K-step.
// Double-buffered LDS; tile k+1 staged with global_load_async_to_lds_b128
// (ASYNCcnt, in-order completion) while tile k is consumed. Steady-state
// loop is branch-free (last iteration peeled). FULL_N=true removes the
// N-bounds guard (Nc multiple of BN). NT=true streams C with non-temporal
// stores (logits are never re-read; keep them out of the 192MB L2).
// ---------------------------------------------------------------------------
#define BM 128
#define BN 256
#define BK 32
#define LDSP 48   // LDS row stride in ushorts: 96B, 16B-aligned for ds_load_b128

template<bool FULL_N, bool NT>
__global__ __launch_bounds__(256)
void k_gemm_bf16(const unsigned short* __restrict__ A,
                 const unsigned short* __restrict__ W,
                 const float* __restrict__ bias,
                 float* __restrict__ C, int ldc,
                 int M, int Nc, int K)
{
    __shared__ unsigned short As[2][BM][LDSP];   // 2 x 12 KB
    __shared__ unsigned short Bs[2][BN][LDSP];   // 2 x 24 KB

    const int tid  = threadIdx.x;
    const int lane = tid & 31;
    const int wave = tid >> 5;        // 0..7
    const int wm   = wave >> 2;       // 0..1  (M dir)
    const int wn   = wave & 3;        // 0..3  (N dir)
    const int lm   = lane & 15;
    const int kh   = lane >> 4;       // 0/1 : K-half selector per ISA layout

    const int m0 = blockIdx.y * BM;
    const int n0 = blockIdx.x * BN;

    // stage one K-tile (6 async b128 per thread) into buffer `buf`
    auto stage = [&](int k0, int buf) {
        #pragma unroll
        for (int it = 0; it < 2; ++it) {          // A: 128x32 bf16, 512 chunks
            int vid = tid + it * 256;
            int row = vid >> 2;
            int col = (vid & 3) << 3;
            async_copy_b128(&As[buf][row][col],
                            &A[(size_t)(m0 + row) * K + k0 + col]);
        }
        #pragma unroll
        for (int it = 0; it < 4; ++it) {          // B: 256x32 bf16, 1024 chunks
            int vid = tid + it * 256;
            int row = vid >> 2;
            int col = (vid & 3) << 3;
            if (FULL_N) {
                async_copy_b128(&Bs[buf][row][col],
                                &W[(size_t)(n0 + row) * K + k0 + col]);
            } else {
                int ng = n0 + row;
                if (ng < Nc)
                    async_copy_b128(&Bs[buf][row][col],
                                    &W[(size_t)ng * K + k0 + col]);
                else
                    *(uint4*)&Bs[buf][row][col] = make_uint4(0u, 0u, 0u, 0u);
            }
        }
    };

    v8f acc[4][4] = {};

    // consume one staged tile from buffer `p`
    auto compute = [&](int p) {
        FragBF16 fa[4], fb[4];
        #pragma unroll
        for (int i = 0; i < 4; ++i) {
            int m = wm * 64 + i * 16 + lm;
            fa[i].q[0] = *(const uint4*)&As[p][m][kh * 8];
            fa[i].q[1] = *(const uint4*)&As[p][m][kh * 8 + 16];
        }
        #pragma unroll
        for (int j = 0; j < 4; ++j) {
            int n = wn * 64 + j * 16 + lm;
            fb[j].q[0] = *(const uint4*)&Bs[p][n][kh * 8];
            fb[j].q[1] = *(const uint4*)&Bs[p][n][kh * 8 + 16];
        }
        #pragma unroll
        for (int i = 0; i < 4; ++i)
            #pragma unroll
            for (int j = 0; j < 4; ++j)
                acc[i][j] = __builtin_amdgcn_wmma_f32_16x16x32_bf16(
                    false, fa[i].v, false, fb[j].v,
                    (short)0, acc[i][j], false, false);
    };

    stage(0, 0);
    int p = 0;
    // steady state: branch-free body, next tile in flight during compute
    for (int k0 = 0; k0 < K - BK; k0 += BK) {
        stage(k0 + BK, p ^ 1);
        if (FULL_N) wait_async6(); else wait_async0();
        __syncthreads();
        compute(p);
        __syncthreads();
        p ^= 1;
    }
    // epilogue: last tile
    wait_async0();
    __syncthreads();
    compute(p);
    __syncthreads();

    // ---- store C (+bias). C/D layout: lane<16 -> M=r, lane>=16 -> M=r+8 ----
    #pragma unroll
    for (int j = 0; j < 4; ++j) {
        int gn = n0 + wn * 64 + j * 16 + lm;
        if (!FULL_N && gn >= Nc) continue;
        float bv = bias ? bias[gn] : 0.0f;
        #pragma unroll
        for (int i = 0; i < 4; ++i) {
            #pragma unroll
            for (int r = 0; r < 8; ++r) {
                int gm = m0 + wm * 64 + i * 16 + (kh ? r + 8 : r);
                float val = acc[i][j][r] + bv;
                if (NT)
                    __builtin_nontemporal_store(val, &C[(size_t)gm * ldc + gn]);
                else
                    C[(size_t)gm * ldc + gn] = val;
            }
        }
    }
}

// ---------------------------------------------------------------------------
// f32 -> bf16 bulk convert (for weights), float4 -> 2 dwords per thread
// ---------------------------------------------------------------------------
__global__ void k_cvt_bf16(const float* __restrict__ src,
                           unsigned short* __restrict__ dst, int n4)
{
    int i = blockIdx.x * blockDim.x + threadIdx.x;
    if (i >= n4) return;
    float4 v = ((const float4*)src)[i];
    ((uint2*)dst)[i] = make_uint2(pack_bf16(v.x, v.y), pack_bf16(v.z, v.w));
}

// ---------------------------------------------------------------------------
// Embedding gather: x/xb[row, d] = emb[ids[row], d]
// ---------------------------------------------------------------------------
__global__ void k_embed(const int* __restrict__ ids,
                        const float* __restrict__ emb,
                        float* __restrict__ x,
                        unsigned short* __restrict__ xb)
{
    int i = blockIdx.x * blockDim.x + threadIdx.x;
    if (i >= BL * DMODEL) return;
    int row = i >> 10;          // / DMODEL
    int d   = i & (DMODEL - 1);
    float v = emb[(size_t)ids[row] * DMODEL + d];
    x[i]  = v;
    xb[i] = to_bf16(v);
}

// ---------------------------------------------------------------------------
// Depthwise conv1d (k=4, pad=2, take first L) + SiLU -> xa (f32) and xab (bf16)
// xs = first D columns of xz (row stride 2D).
// ---------------------------------------------------------------------------
__global__ void k_conv_silu(const float* __restrict__ xz,
                            const float* __restrict__ cw,
                            const float* __restrict__ cb,
                            float* __restrict__ xa,
                            unsigned short* __restrict__ xab)
{
    int i = blockIdx.x * blockDim.x + threadIdx.x;
    if (i >= BL * DMODEL) return;
    int d   = i & (DMODEL - 1);
    int row = i >> 10;              // b*SEQ + l
    int l   = row & (SEQ - 1);
    int b   = row >> 11;            // / SEQ
    const float* xs = xz + (size_t)b * SEQ * 2 * DMODEL;
    float accv = cb[d];
    #pragma unroll
    for (int j = 0; j < 4; ++j) {
        int li = l + j - 2;
        if (li >= 0 && li < SEQ)
            accv = fmaf(cw[d * 4 + j], xs[(size_t)li * 2 * DMODEL + d], accv);
    }
    float v = silu(accv);
    xa[i]  = v;
    xab[i] = to_bf16(v);
}

// ---------------------------------------------------------------------------
// Selective scan. One wave32 per (b,d) row, lane = state index n.
//   s = clip(s*A + B[l,n]*xa[l,d], +/-10);  y[l,d] = sum_n C[l,n]*s
// Cross-lane reduce via shfl_xor butterfly (wave32).
// ---------------------------------------------------------------------------
__global__ __launch_bounds__(256)
void k_scan(const float* __restrict__ xa,
            const float* __restrict__ Bm,
            const float* __restrict__ Cm,
            const float* __restrict__ alog,
            float* __restrict__ y)
{
    int wid  = blockIdx.x * 8 + (threadIdx.x >> 5);   // 0 .. B*D-1
    int lane = threadIdx.x & 31;
    int b = wid >> 10;              // / DMODEL
    int d = wid & (DMODEL - 1);

    float al = alog[d * NSTATE + lane];
    al = fminf(fmaxf(al, -5.0f), 2.0f);
    const float Acoef = -expf(al);

    const float* xap = xa + (size_t)b * SEQ * DMODEL + d;
    const float* bp  = Bm + (size_t)b * SEQ * NSTATE + lane;
    const float* cp  = Cm + (size_t)b * SEQ * NSTATE + lane;
    float*       yp  = y  + (size_t)b * SEQ * DMODEL + d;

    float s = 0.0f;
    for (int l = 0; l < SEQ; ++l) {
        float xt = xap[(size_t)l * DMODEL];
        float bt = bp[(size_t)l * NSTATE];
        float ct = cp[(size_t)l * NSTATE];
        s = fminf(fmaxf(fmaf(bt, xt, s * Acoef), -10.0f), 10.0f);
        float p = ct * s;
        p += __shfl_xor(p, 16, 32);
        p += __shfl_xor(p, 8, 32);
        p += __shfl_xor(p, 4, 32);
        p += __shfl_xor(p, 2, 32);
        p += __shfl_xor(p, 1, 32);
        if (lane == 0) yp[(size_t)l * DMODEL] = p;
    }
}

// ---------------------------------------------------------------------------
// Gate: yb = bf16( y * silu(z) ), z = second D columns of xz
// ---------------------------------------------------------------------------
__global__ void k_gate(const float* __restrict__ y,
                       const float* __restrict__ xz,
                       unsigned short* __restrict__ yb)
{
    int i = blockIdx.x * blockDim.x + threadIdx.x;
    if (i >= BL * DMODEL) return;
    int row = i >> 10;
    int d   = i & (DMODEL - 1);
    float z = xz[(size_t)row * 2 * DMODEL + DMODEL + d];
    yb[i] = to_bf16(y[i] * silu(z));
}

// ---------------------------------------------------------------------------
// Residual add + LayerNorm, one block (256 thr) per token row; writes x, xb.
// ---------------------------------------------------------------------------
__global__ __launch_bounds__(256)
void k_add_ln(const float* __restrict__ o, float* __restrict__ x,
              unsigned short* __restrict__ xb,
              const float* __restrict__ g, const float* __restrict__ be)
{
    __shared__ float red[256];
    const int row = blockIdx.x;
    const int tid = threadIdx.x;
    float t[4];
    float sum = 0.0f;
    #pragma unroll
    for (int c = 0; c < 4; ++c) {
        int d = tid + c * 256;
        t[c] = o[(size_t)row * DMODEL + d] + x[(size_t)row * DMODEL + d];
        sum += t[c];
    }
    red[tid] = sum; __syncthreads();
    for (int off = 128; off > 0; off >>= 1) {
        if (tid < off) red[tid] += red[tid + off];
        __syncthreads();
    }
    const float mu = red[0] / DMODEL;
    __syncthreads();
    float vs = 0.0f;
    #pragma unroll
    for (int c = 0; c < 4; ++c) { float dd = t[c] - mu; vs += dd * dd; }
    red[tid] = vs; __syncthreads();
    for (int off = 128; off > 0; off >>= 1) {
        if (tid < off) red[tid] += red[tid + off];
        __syncthreads();
    }
    const float inv = rsqrtf(red[0] / DMODEL + 1e-5f);
    #pragma unroll
    for (int c = 0; c < 4; ++c) {
        int d = tid + c * 256;
        float v = (t[c] - mu) * inv * g[d] + be[d];
        x[(size_t)row * DMODEL + d]  = v;
        xb[(size_t)row * DMODEL + d] = to_bf16(v);
    }
}

// ---------------------------------------------------------------------------
// Launcher
// ---------------------------------------------------------------------------
extern "C" void kernel_launch(void* const* d_in, const int* in_sizes, int n_in,
                              void* d_out, int out_size, void* d_ws, size_t ws_size,
                              hipStream_t stream)
{
    const int*   ids    = (const int*)  d_in[0];
    const float* emb    = (const float*)d_in[1];
    const float* in_w   = (const float*)d_in[2];
    const float* in_b   = (const float*)d_in[3];
    const float* conv_w = (const float*)d_in[4];
    const float* conv_b = (const float*)d_in[5];
    const float* A_log  = (const float*)d_in[6];
    const float* bp_w   = (const float*)d_in[7];
    const float* bp_b   = (const float*)d_in[8];
    const float* cp_w   = (const float*)d_in[9];
    const float* cp_b   = (const float*)d_in[10];
    const float* out_w  = (const float*)d_in[11];
    const float* out_b  = (const float*)d_in[12];
    const float* ln_g   = (const float*)d_in[13];
    const float* ln_b   = (const float*)d_in[14];
    const float* head_w = (const float*)d_in[15];
    float* logits = (float*)d_out;

    // workspace layout
    char* p = (char*)d_ws;
    float* x   = (float*)p;  p += (size_t)BL * DMODEL * 4;       // f32 residual
    float* xz  = (float*)p;  p += (size_t)BL * 2 * DMODEL * 4;
    float* xa  = (float*)p;  p += (size_t)BL * DMODEL * 4;
    float* y   = (float*)p;  p += (size_t)BL * DMODEL * 4;
    float* o   = (float*)p;  p += (size_t)BL * DMODEL * 4;
    float* Bm  = (float*)p;  p += (size_t)BL * NSTATE * 4;
    float* Cm  = (float*)p;  p += (size_t)BL * NSTATE * 4;
    unsigned short* xb  = (unsigned short*)p; p += (size_t)BL * DMODEL * 2;
    unsigned short* xab = (unsigned short*)p; p += (size_t)BL * DMODEL * 2;
    unsigned short* yb  = (unsigned short*)p; p += (size_t)BL * DMODEL * 2;
    unsigned short* wbf = (unsigned short*)p; p += (size_t)VOCAB * DMODEL * 2; // weight arena

    const int EL = BL * DMODEL;
    const dim3 eb(256), eg((EL + 255) / 256);

    k_embed<<<eg, eb, 0, stream>>>(ids, emb, x, xb);

    for (int l = 0; l < NLAYER; ++l) {
        const float* iw = in_w  + (size_t)l * 2 * DMODEL * DMODEL;
        const float* ib = in_b  + (size_t)l * 2 * DMODEL;
        const float* cw = conv_w+ (size_t)l * DMODEL * 4;
        const float* cb = conv_b+ (size_t)l * DMODEL;
        const float* al = A_log + (size_t)l * DMODEL * NSTATE;
        const float* bw = bp_w  + (size_t)l * NSTATE * DMODEL;
        const float* bb = bp_b  + (size_t)l * NSTATE;
        const float* c2w= cp_w  + (size_t)l * NSTATE * DMODEL;
        const float* c2b= cp_b  + (size_t)l * NSTATE;
        const float* ow = out_w + (size_t)l * DMODEL * DMODEL;
        const float* ob = out_b + (size_t)l * DMODEL;
        const float* g  = ln_g  + (size_t)l * DMODEL;
        const float* be = ln_b  + (size_t)l * DMODEL;

        // in_proj: xz = x @ iw^T + ib            [4096 x 2048]
        {
            int n4 = 2 * DMODEL * DMODEL / 4;
            k_cvt_bf16<<<(n4 + 255) / 256, 256, 0, stream>>>(iw, wbf, n4);
            k_gemm_bf16<true, false><<<dim3(2 * DMODEL / BN, BL / BM), 256, 0, stream>>>(
                xb, wbf, ib, xz, 2 * DMODEL, BL, 2 * DMODEL, DMODEL);
        }

        // depthwise conv + silu -> xa, xab
        k_conv_silu<<<eg, eb, 0, stream>>>(xz, cw, cb, xa, xab);

        // B/C projections                        [4096 x 32]
        {
            int n4 = NSTATE * DMODEL / 4;
            k_cvt_bf16<<<(n4 + 255) / 256, 256, 0, stream>>>(bw, wbf, n4);
            k_gemm_bf16<false, false><<<dim3(1, BL / BM), 256, 0, stream>>>(
                xab, wbf, bb, Bm, NSTATE, BL, NSTATE, DMODEL);
            k_cvt_bf16<<<(n4 + 255) / 256, 256, 0, stream>>>(c2w, wbf, n4);
            k_gemm_bf16<false, false><<<dim3(1, BL / BM), 256, 0, stream>>>(
                xab, wbf, c2b, Cm, NSTATE, BL, NSTATE, DMODEL);
        }

        // selective scan (one wave per (b,d))
        k_scan<<<(BSZ * DMODEL) / 8, 256, 0, stream>>>(xa, Bm, Cm, al, y);

        // gate -> yb (bf16)
        k_gate<<<eg, eb, 0, stream>>>(y, xz, yb);

        // out_proj: o = yb @ ow^T + ob           [4096 x 1024]
        {
            int n4 = DMODEL * DMODEL / 4;
            k_cvt_bf16<<<(n4 + 255) / 256, 256, 0, stream>>>(ow, wbf, n4);
            k_gemm_bf16<true, false><<<dim3(DMODEL / BN, BL / BM), 256, 0, stream>>>(
                yb, wbf, ob, o, DMODEL, BL, DMODEL, DMODEL);
        }

        // residual + layernorm -> x, xb
        k_add_ln<<<BL, 256, 0, stream>>>(o, x, xb, g, be);
    }

    // LM head: logits = xb @ head_w^T            [4096 x 32000]
    // NT stores: 524MB of stream-out, never re-read -> keep out of L2
    {
        int n4 = VOCAB * DMODEL / 4;
        k_cvt_bf16<<<(n4 + 255) / 256, 256, 0, stream>>>(head_w, wbf, n4);
        k_gemm_bf16<true, true><<<dim3(VOCAB / BN, BL / BM), 256, 0, stream>>>(
            xb, wbf, nullptr, logits, VOCAB, BL, VOCAB, DMODEL);
    }
}
